// WaveNet_6451040878926
// MI455X (gfx1250) — matrix operand order repaired and checked by reading, not compiled, not run on hardware
//
#include <hip/hip_runtime.h>
#include <hip/hip_bf16.h>

#define CCH   64
#define TLEN  32768
#define BATCH 8
#define NLAY  20
#define TN    64   // time tile per workgroup

typedef __attribute__((ext_vector_type(16))) __bf16 v16bf;
typedef __attribute__((ext_vector_type(8)))  __bf16 v8bf;
typedef __attribute__((ext_vector_type(4)))  __bf16 v4bf;
typedef __attribute__((ext_vector_type(8)))  float  v8f;

union BF16Frag { v16bf v; v8bf h[2]; };

__device__ __forceinline__ float sigmoidf_(float x) { return 1.f / (1.f + __expf(-x)); }

// ---------------------------------------------------------------------------
// Weight prep: f32 -> bf16, reorder w_h [L][co][ci][k] -> [L][k][co][ci]
// w_r is [L][co][ci][1] -> [L][co][ci] (same order)
// ---------------------------------------------------------------------------
__global__ void prep_weights(const float* __restrict__ wh, const float* __restrict__ wr,
                             __bf16* __restrict__ whb, __bf16* __restrict__ wrb) {
    int idx = blockIdx.x * blockDim.x + threadIdx.x;
    const int nh = NLAY * 3 * 128 * 64;
    if (idx < nh) {
        int ci = idx & 63;
        int co = (idx >> 6) & 127;
        int lk = idx >> 13;          // l*3 + k
        int k  = lk % 3;
        int l  = lk / 3;
        whb[idx] = (__bf16)wh[((l * 128 + co) * 64 + ci) * 3 + k];
    }
    const int nr = NLAY * 64 * 64;
    if (idx < nr) {
        wrb[idx] = (__bf16)wr[idx];
    }
}

// ---------------------------------------------------------------------------
// Input 1x1 conv: resF/resH[b][t][c] = w_in[c]*x[b][t] + b_in[c]
// ---------------------------------------------------------------------------
__global__ void input_conv(const float* __restrict__ x, const float* __restrict__ w_in,
                           const float* __restrict__ b_in,
                           float* __restrict__ resF, __bf16* __restrict__ resH) {
    int idx = blockIdx.x * blockDim.x + threadIdx.x;   // B*T*64 threads
    int c  = idx & 63;
    int bt = idx >> 6;
    float v = w_in[c] * x[bt] + b_in[c];
    resF[idx] = v;
    resH[idx] = (__bf16)v;
}

// out[b][t] = b_mix  (skip accumulation target)
__global__ void init_out(float* __restrict__ out, const float* __restrict__ b_mix) {
    int idx = blockIdx.x * blockDim.x + threadIdx.x;
    if (idx < BATCH * TLEN) out[idx] = b_mix[0];
}

// ---------------------------------------------------------------------------
// One WaveNet layer, fully fused:
//   h = W_h(*3 dilated taps*) @ res_in            (WMMA bf16, f32 acc)
//   gated = tanh(h[:64]+bh[:64]) * sigmoid(h[64:]+bh[64:])
//   skip  += w_mix_layer . gated                  (fused mix conv)
//   res_out = W_r @ gated + b_r + res_in          (WMMA)
// x staging uses GLOBAL_LOAD_ASYNC_TO_LDS_B128 from the bf16 shadow stream.
// ---------------------------------------------------------------------------
__launch_bounds__(256)
__global__ void wavenet_layer(const float* __restrict__ resInF,
                              const __bf16* __restrict__ resInH,
                              float* __restrict__ resOutF,
                              __bf16* __restrict__ resOutH,
                              const __bf16* __restrict__ whb,   // [3][128][64]
                              const __bf16* __restrict__ wrb,   // [64][64]
                              const float* __restrict__ bh,     // [128]
                              const float* __restrict__ br,     // [64]
                              const float* __restrict__ wmix,   // [64]
                              float* __restrict__ skip,         // [B*T] accum
                              int dil) {
    // LDS: 3*64*72 bf16 (27648B) staging | 64*132 f32 h (33792B) = 61440B.
    // gated (64*72 bf16) and psum (64*4 f32) alias the staging region, which
    // is dead after the h-GEMM barrier.
    __shared__ alignas(16) char smem[27648 + 33792];
    __bf16 (*xs)[TN][72] = (__bf16(*)[TN][72])smem;            // [3][64][72]
    float  (*hbuf)[132]  = (float(*)[132])(smem + 27648);      // [64][132]
    __bf16 (*gbuf)[72]   = (__bf16(*)[72])smem;                // [64][72] (alias xs[0])
    float  (*psum)[4]    = (float(*)[4])(smem + 9216);         // [64][4]  (alias xs[1])

    const int tid  = threadIdx.x;
    const int b    = blockIdx.y;
    const int t0   = blockIdx.x * TN;
    const long baseBT = (long)b * TLEN;

    // ---- Stage the 3 dilated-tap time ranges into LDS (async, no VGPR data)
    {
        const unsigned long long sbase =
            (unsigned long long)(resInH + baseBT * 64);   // bf16 base for this batch
        // 1536 16-byte chunks: [stage k][row][8 chunks of 8 bf16]
        #pragma unroll
        for (int it = 0; it < 6; ++it) {
            int chunk = tid + it * 256;
            int k   = chunk >> 9;            // 0..2
            int rem = chunk & 511;
            int row = rem >> 3;              // 0..63
            int cc  = (rem & 7) << 3;        // element offset 0,8,...,56
            int t   = t0 + row - k * dil;
            unsigned ldsa = (unsigned)(size_t)(&xs[k][row][cc]);
            if (t >= 0) {
                unsigned voff = (unsigned)((t * 64 + cc) * 2);
                asm volatile("global_load_async_to_lds_b128 %0, %1, %2"
                             :: "v"(ldsa), "v"(voff), "s"(sbase) : "memory");
            } else {
                v8bf z = {};
                *(v8bf*)&xs[k][row][cc] = z;
            }
        }
        asm volatile("s_wait_asynccnt 0" ::: "memory");
    }
    __syncthreads();

    const int wave = tid >> 5;
    const int lane = tid & 31;
    const int half = lane >> 4;   // which 16-lane group
    const int ln   = lane & 15;

    // ---- h GEMM: wave owns M-tile (16 output channels of 128) -------------
    // A frag (16x32 bf16): lane(half,m): elems 0-7 -> K = 8*half + j,
    //                       elems 8-15 -> K = 16 + 8*half + (j-8)
    BF16Frag afr[3][2];
    {
        int co = wave * 16 + ln;
        #pragma unroll
        for (int k = 0; k < 3; ++k)
            #pragma unroll
            for (int ks = 0; ks < 2; ++ks) {
                const __bf16* p = whb + ((k * 128 + co) * 64) + ks * 32 + 8 * half;
                afr[k][ks].h[0] = *(const v8bf*)(p);
                afr[k][ks].h[1] = *(const v8bf*)(p + 16);
            }
    }

    v8f acc[4] = {};
    #pragma unroll
    for (int ns = 0; ns < 4; ++ns) {
        #pragma unroll
        for (int k = 0; k < 3; ++k) {
            #pragma unroll
            for (int ks = 0; ks < 2; ++ks) {
                // B frag (32x16 bf16): lane = column n; elems 0-7 -> K = 8*half + j, elems 8-15 -> K = 16 + 8*half + (j-8)
                BF16Frag bf;
                const __bf16* p = &xs[k][ns * 16 + ln][ks * 32 + 8 * half];
                bf.h[0] = *(const v8bf*)p;
                bf.h[1] = *(const v8bf*)(p + 16);
                acc[ns] = __builtin_amdgcn_wmma_f32_16x16x32_bf16(
                    false, afr[k][ks].v, false, bf.v, (short)0, acc[ns], false, false);
            }
        }
    }
    // D layout: vgpr r, lane(half,n) -> (m = r + 8*half, n). Store h[t][ch].
    #pragma unroll
    for (int ns = 0; ns < 4; ++ns) {
        float* hp = &hbuf[ns * 16 + ln][wave * 16 + 8 * half];
        *(float4*)hp       = make_float4(acc[ns][0], acc[ns][1], acc[ns][2], acc[ns][3]);
        *(float4*)(hp + 4) = make_float4(acc[ns][4], acc[ns][5], acc[ns][6], acc[ns][7]);
    }
    __syncthreads();

    // ---- Gating + fused skip-mix partial sums (writes alias dead xs) ------
    {
        int t  = tid & 63;
        int cg = tid >> 6;        // 0..3, 16 channels each
        float s = 0.f;
        #pragma unroll
        for (int i = 0; i < 16; ++i) {
            int c   = cg * 16 + i;
            float a = hbuf[t][c]      + bh[c];
            float g = hbuf[t][c + 64] + bh[c + 64];
            float gv = tanhf(a) * sigmoidf_(g);
            gbuf[t][c] = (__bf16)gv;
            s += wmix[c] * gv;
        }
        psum[t][cg] = s;
    }
    __syncthreads();
    if (tid < 64) {
        int t = tid;
        float s = psum[t][0] + psum[t][1] + psum[t][2] + psum[t][3];
        skip[baseBT + t0 + t] += s;
    }

    // ---- Residual 1x1 GEMM (M=64): wave -> (M-tile = wave&3, N-half = wave>>2)
    {
        const int mt = wave & 3;
        const int nh = wave >> 2;
        BF16Frag ar[2];
        int co = mt * 16 + ln;
        #pragma unroll
        for (int ks = 0; ks < 2; ++ks) {
            const __bf16* p = wrb + co * 64 + ks * 32 + 8 * half;
            ar[ks].h[0] = *(const v8bf*)p;
            ar[ks].h[1] = *(const v8bf*)(p + 16);
        }
        #pragma unroll
        for (int j = 0; j < 2; ++j) {
            int ns = nh * 2 + j;
            v8f racc = {};
            #pragma unroll
            for (int ks = 0; ks < 2; ++ks) {
                BF16Frag bf;
                const __bf16* p = &gbuf[ns * 16 + ln][ks * 32 + 8 * half];
                bf.h[0] = *(const v8bf*)p;
                bf.h[1] = *(const v8bf*)(p + 16);
                racc = __builtin_amdgcn_wmma_f32_16x16x32_bf16(
                    false, ar[ks].v, false, bf.v, (short)0, racc, false, false);
            }
            int t  = t0 + ns * 16 + ln;
            int c0 = mt * 16 + 8 * half;
            const float* rip = resInF  + (baseBT + t) * 64 + c0;
            float*       rop = resOutF + (baseBT + t) * 64 + c0;
            float4 r0 = *(const float4*)rip;
            float4 r1 = *(const float4*)(rip + 4);
            float o[8];
            o[0] = racc[0] + r0.x + br[c0 + 0];
            o[1] = racc[1] + r0.y + br[c0 + 1];
            o[2] = racc[2] + r0.z + br[c0 + 2];
            o[3] = racc[3] + r0.w + br[c0 + 3];
            o[4] = racc[4] + r1.x + br[c0 + 4];
            o[5] = racc[5] + r1.y + br[c0 + 5];
            o[6] = racc[6] + r1.z + br[c0 + 6];
            o[7] = racc[7] + r1.w + br[c0 + 7];
            *(float4*)rop       = make_float4(o[0], o[1], o[2], o[3]);
            *(float4*)(rop + 4) = make_float4(o[4], o[5], o[6], o[7]);
            // bf16 shadow for the next layer's async staging
            v8bf ob;
            #pragma unroll
            for (int q = 0; q < 8; ++q) ob[q] = (__bf16)o[q];
            *(v8bf*)(resOutH + (baseBT + t) * 64 + c0) = ob;
        }
    }
}

// ---------------------------------------------------------------------------
extern "C" void kernel_launch(void* const* d_in, const int* in_sizes, int n_in,
                              void* d_out, int out_size, void* d_ws, size_t ws_size,
                              hipStream_t stream) {
    const float* x     = (const float*)d_in[0];
    const float* w_in  = (const float*)d_in[1];
    const float* b_in  = (const float*)d_in[2];
    const float* w_h   = (const float*)d_in[3];
    const float* b_h   = (const float*)d_in[4];
    const float* w_r   = (const float*)d_in[5];
    const float* b_r   = (const float*)d_in[6];
    const float* w_mix = (const float*)d_in[7];
    const float* b_mix = (const float*)d_in[8];
    float* out = (float*)d_out;

    const size_t resElems = (size_t)BATCH * TLEN * 64;
    float*  resAF = (float*)d_ws;
    float*  resBF = resAF + resElems;
    __bf16* resAH = (__bf16*)(resBF + resElems);
    __bf16* resBH = resAH + resElems;
    __bf16* whb   = resBH + resElems;
    __bf16* wrb   = whb + (size_t)NLAY * 3 * 128 * 64;

    prep_weights<<<(NLAY * 3 * 128 * 64 + 255) / 256, 256, 0, stream>>>(w_h, w_r, whb, wrb);
    init_out<<<(BATCH * TLEN + 255) / 256, 256, 0, stream>>>(out, b_mix);
    input_conv<<<(BATCH * TLEN * 64) / 256, 256, 0, stream>>>(x, w_in, b_in, resAF, resAH);

    const int dils[NLAY] = {1, 2, 4, 8, 16, 32, 64, 128, 256, 512,
                            1, 2, 4, 8, 16, 32, 64, 128, 256, 512};
    float*  rinF  = resAF;
    float*  routF = resBF;
    __bf16* rinH  = resAH;
    __bf16* routH = resBH;
    dim3 grid(TLEN / TN, BATCH);
    for (int l = 0; l < NLAY; ++l) {
        wavenet_layer<<<grid, 256, 0, stream>>>(
            rinF, rinH, routF, routH,
            whb + (size_t)l * 3 * 128 * 64,
            wrb + (size_t)l * 64 * 64,
            b_h + l * 128, b_r + l * 64, w_mix + l * 64,
            out, dils[l]);
        float*  tf = rinF; rinF = routF; routF = tf;
        __bf16* th = rinH; rinH = routH; routH = th;
    }
}
